// SEGNN_41326175322391
// MI455X (gfx1250) — matrix operand order repaired and checked by reading, not compiled, not run on hardware
//
#include <hip/hip_runtime.h>
#include <math.h>

#define N_NODES 20000
#define N_EDGES 320000
#define STEPS 3

typedef _Float16 half_t;
typedef half_t v16h __attribute__((ext_vector_type(16)));
typedef half_t h2   __attribute__((ext_vector_type(2)));
typedef float  v8f  __attribute__((ext_vector_type(8)));

__device__ __forceinline__ float sigmoid_f(float x) { return 1.0f / (1.0f + expf(-x)); }
__device__ __forceinline__ float gelu_f(float x) {
    float u = 0.7978845608028654f * (x + 0.044715f * x * x * x);
    return 0.5f * x * (1.0f + tanhf(u));
}

__device__ __forceinline__ v8f wmma16(v16h a, v16h b, v8f c) {
    // v_wmma_f32_16x16x32_f16 : D = A(16x32 f16) * B(32x16 f16) + C(f32)
    return __builtin_amdgcn_wmma_f32_16x16x32_f16(false, a, false, b, (short)0, c,
                                                  false, false);
}

// A fragment: 16-bit A-matrix 16x32 layout (ISA 7.12.2).
// lane L: M = L&15 ; kbase = 8*(L>=16) ; K(e) = kbase + e + (e>=8 ? 8 : 0)
// Two contiguous 8-half runs per lane -> compiler emits 2x ds_load_b128.
__device__ __forceinline__ v16h load_A_frag(const half_t* tile, int lane, int kOff) {
    int M  = lane & 15;
    int kb = (lane >> 4) * 8;
    v16h a;
#pragma unroll
    for (int e = 0; e < 16; ++e) {
        int k = kb + e + ((e >= 8) ? 8 : 0);
        a[e] = tile[M * 64 + kOff + k];
    }
    return a;
}

// B fragment: pre-swizzled in LDS (fragment-major), one aligned 32B load per lane.
__device__ __forceinline__ v16h load_B_pre(const half_t* frag, int lane) {
    return *(const v16h*)(frag + lane * 16);
}

// ---------------------------------------------------------------------------
// Utility kernels
// ---------------------------------------------------------------------------
__global__ void init_state(const float* __restrict__ s_in, const float* __restrict__ v_in,
                           float* __restrict__ s_cur, float* __restrict__ v_cur, int N) {
    int i = blockIdx.x * blockDim.x + threadIdx.x;
    if (i < N * 16) s_cur[i] = s_in[i];
    if (i < N * 3)  v_cur[i] = v_in[i];
}

__global__ void zero_kernel(float* __restrict__ p, int n) {
    int i = blockIdx.x * blockDim.x + threadIdx.x;
    if (i < n) p[i] = 0.0f;
}

__global__ void write_out(const float* __restrict__ s_cur, const float* __restrict__ v_cur,
                          float* __restrict__ out, int N) {
    int n = blockIdx.x * blockDim.x + threadIdx.x;
    if (n >= N) return;
#pragma unroll
    for (int k = 0; k < 16; ++k) out[n * 19 + k] = s_cur[n * 16 + k];
#pragma unroll
    for (int d = 0; d < 3; ++d)  out[n * 19 + 16 + d] = v_cur[n * 3 + d];
}

// ---------------------------------------------------------------------------
// Edge kernel: 4 waves/block, each wave processes one 16-edge WMMA tile.
// B sub-tile numbering in sBfrag (each 512 halves = 32 lanes x 16):
//   0..3 : Ws0 (64x32), sub = kc*2+nc
//   4..5 : Wv0 (64x16), sub = 4+kc
//   6..7 : Ws1 (32x32), sub = 6+nc
//   8    : Wv1 (32x16)
// ---------------------------------------------------------------------------
#define EWAVES 4
#define EPB (EWAVES * 16)

__global__ void __launch_bounds__(EWAVES * 32)
edge_kernel(const float* __restrict__ s_cur, const float* __restrict__ v_cur,
            const int* __restrict__ senders, const int* __restrict__ recvs,
            const float* __restrict__ Ws0, const float* __restrict__ bs0,
            const float* __restrict__ Wv0,
            const float* __restrict__ Ws1, const float* __restrict__ bs1,
            const float* __restrict__ Wv1,
            float* __restrict__ agg_ms, float* __restrict__ agg_mv,
            float* __restrict__ agg_av, float* __restrict__ agg_cnt, int E) {
    __shared__ __attribute__((aligned(32))) half_t sBfrag[9 * 512];
    __shared__ float  sB0[32];
    __shared__ float  sB1[32];
    __shared__ __attribute__((aligned(32))) half_t sA[EWAVES][16 * 64];
    __shared__ float  sRaw[EWAVES][16 * 64];  // per-edge raw data (f32)
    __shared__ float  sS1[EWAVES][16 * 32];   // scalar GEMM result (layer1 then layer2)
    __shared__ float  sV1[EWAVES][16 * 48];   // vector GEMM result [row][comp*16+q]
    __shared__ int    sRecv[EWAVES][16];

    int tid = threadIdx.x;
    __builtin_prefetch(Ws0, 0, 1);
    __builtin_prefetch(Wv0, 0, 1);
    __builtin_prefetch(Ws1, 0, 1);
    __builtin_prefetch(Wv1, 0, 1);

    // ---- stage f16 weights into LDS, pre-swizzled to B-fragment layout ----
    for (int i = tid; i < 9 * 512; i += blockDim.x) {
        int sub  = i >> 9;
        int idx  = i & 511;
        int lane = idx >> 4;
        int e    = idx & 15;
        int NN   = lane & 15;
        int KK   = ((lane >> 4) * 8) + e + ((e >= 8) ? 8 : 0);  // 0..31 within sub-tile
        float val;
        if (sub < 4) {                       // Ws0 [34,32] K-padded to 64
            int k = (sub >> 1) * 32 + KK;
            int n = (sub & 1) * 16 + NN;
            val = (k < 34) ? Ws0[k * 32 + n] : 0.0f;
        } else if (sub < 6) {                // Wv0 [34,16] K-padded to 64
            int k = (sub - 4) * 32 + KK;
            val = (k < 34) ? Wv0[k * 16 + NN] : 0.0f;
        } else if (sub < 8) {                // Ws1 [32,32]
            int n = (sub - 6) * 16 + NN;
            val = Ws1[KK * 32 + n];
        } else {                             // Wv1 [32,16]
            val = Wv1[KK * 16 + NN];
        }
        sBfrag[i] = (half_t)val;
    }
    if (tid < 32) { sB0[tid] = bs0[tid]; sB1[tid] = bs1[tid]; }
    __syncthreads();

    int w    = tid >> 5;
    int lane = tid & 31;
    int lo   = lane & 15, hi = lane >> 4;
    int tile0 = blockIdx.x * EPB + w * 16;

    // ---- phase 1: gather node features, build ts tile [16 x 34->64] ----
    if (lane < 16) {
        int j  = lane;
        int e  = tile0 + j;
        int ei = (e < E) ? e : (E - 1);
        int snd = senders[ei];
        int rcv = recvs[ei];
        sRecv[w][j] = rcv;
        float* rw = &sRaw[w][j * 64];
#pragma unroll
        for (int k = 0; k < 16; ++k) rw[k]      = s_cur[snd * 16 + k];
#pragma unroll
        for (int k = 0; k < 16; ++k) rw[16 + k] = s_cur[rcv * 16 + k];
        float vs0 = v_cur[snd * 3 + 0], vs1 = v_cur[snd * 3 + 1], vs2 = v_cur[snd * 3 + 2];
        float vr0 = v_cur[rcv * 3 + 0], vr1 = v_cur[rcv * 3 + 1], vr2 = v_cur[rcv * 3 + 2];
        float r0 = vs0 - vr0, r1 = vs1 - vr1, r2 = vs2 - vr2;
        float rn = sqrtf(r0 * r0 + r1 * r1 + r2 * r2) + 1e-8f;
        float k3 = 1.7320508075688772f / rn;  // sqrt(3) * rhat
        float a0 = r0 * k3, a1 = r1 * k3, a2 = r2 * k3;
        rw[32] = vs0; rw[33] = vs1; rw[34] = vs2;
        rw[35] = vr0; rw[36] = vr1; rw[37] = vr2;
        rw[38] = a0;  rw[39] = a1;  rw[40] = a2;
        float dsend = vs0 * a0 + vs1 * a1 + vs2 * a2;
        float drecv = vr0 * a0 + vr1 * a1 + vr2 * a2;
        h2* tr2 = (h2*)&sA[w][j * 64];
#pragma unroll
        for (int k = 0; k < 16; ++k) {
            h2 t = {(half_t)rw[2 * k], (half_t)rw[2 * k + 1]};
            tr2[k] = t;
        }
        h2 td = {(half_t)dsend, (half_t)drecv};
        tr2[16] = td;
        h2 z = {(half_t)0.0f, (half_t)0.0f};
#pragma unroll
        for (int k = 17; k < 32; ++k) tr2[k] = z;
    }
    __syncthreads();

    // ---- phase 2: layer-1 scalar GEMM  [16x64] x [64x32] via 4 WMMAs ----
    {
        v16h A0 = load_A_frag(&sA[w][0], lane, 0);
        v16h A1 = load_A_frag(&sA[w][0], lane, 32);
#pragma unroll
        for (int nc = 0; nc < 2; ++nc) {
            v8f acc = {};
            acc = wmma16(A0, load_B_pre(&sBfrag[(0 + nc) * 512], lane), acc);
            acc = wmma16(A1, load_B_pre(&sBfrag[(2 + nc) * 512], lane), acc);
            int col  = nc * 16 + lo;
            float bb = sB0[col];
#pragma unroll
            for (int i = 0; i < 8; ++i) sS1[w][(i + 8 * hi) * 32 + col] = acc[i] + bb;
        }
    }
    __syncthreads();

    // ---- phase 3: layer-1 vector GEMM, per component (3 x 2 WMMAs) ----
    for (int comp = 0; comp < 3; ++comp) {
        if (lane < 16) {
            int j = lane;
            float* rw = &sRaw[w][j * 64];
            float avc = rw[38 + comp];
            h2* tr2 = (h2*)&sA[w][j * 64];
#pragma unroll
            for (int k = 0; k < 16; ++k) {
                h2 t = {(half_t)(rw[2 * k] * avc), (half_t)(rw[2 * k + 1] * avc)};
                tr2[k] = t;
            }
            h2 td = {(half_t)rw[32 + comp], (half_t)rw[35 + comp]};
            tr2[16] = td;
            h2 z = {(half_t)0.0f, (half_t)0.0f};
#pragma unroll
            for (int k = 17; k < 32; ++k) tr2[k] = z;
        }
        __syncthreads();
        v16h A0 = load_A_frag(&sA[w][0], lane, 0);
        v16h A1 = load_A_frag(&sA[w][0], lane, 32);
        v8f acc = {};
        acc = wmma16(A0, load_B_pre(&sBfrag[4 * 512], lane), acc);
        acc = wmma16(A1, load_B_pre(&sBfrag[5 * 512], lane), acc);
#pragma unroll
        for (int i = 0; i < 8; ++i)
            sV1[w][(i + 8 * hi) * 48 + comp * 16 + lo] = acc[i];
        __syncthreads();
    }

    // ---- phase 4: gate layer-1, build layer-2 ts tile [16 x 32] ----
    if (lane < 16) {
        int j = lane;
        float* rw = &sRaw[w][j * 64];
        float* s1 = &sS1[w][j * 32];
        float* v1 = &sV1[w][j * 48];
        float g[16];
#pragma unroll
        for (int q = 0; q < 16; ++q) g[q] = sigmoid_f(s1[q]);
#pragma unroll
        for (int comp = 0; comp < 3; ++comp)
#pragma unroll
            for (int q = 0; q < 16; ++q) v1[comp * 16 + q] *= g[q];  // mv = v1 * gate
#pragma unroll
        for (int k = 0; k < 16; ++k) rw[41 + k] = gelu_f(s1[16 + k]);  // ms
        h2* tr2 = (h2*)&sA[w][j * 64];
#pragma unroll
        for (int k = 0; k < 8; ++k) {
            h2 t = {(half_t)rw[41 + 2 * k], (half_t)rw[42 + 2 * k]};
            tr2[k] = t;
        }
#pragma unroll
        for (int q = 0; q < 16; q += 2) {
            float d0 = v1[q] * rw[38] + v1[16 + q] * rw[39] + v1[32 + q] * rw[40];
            float d1 = v1[q + 1] * rw[38] + v1[17 + q] * rw[39] + v1[33 + q] * rw[40];
            h2 t = {(half_t)d0, (half_t)d1};
            tr2[8 + (q >> 1)] = t;  // mv . a_v
        }
    }
    __syncthreads();

    // ---- phase 5: layer-2 scalar GEMM  [16x32] x [32x32] via 2 WMMAs ----
    {
        v16h A = load_A_frag(&sA[w][0], lane, 0);
#pragma unroll
        for (int nc = 0; nc < 2; ++nc) {
            v8f acc = {};
            acc = wmma16(A, load_B_pre(&sBfrag[(6 + nc) * 512], lane), acc);
            int col  = nc * 16 + lo;
            float bb = sB1[col];
#pragma unroll
            for (int i = 0; i < 8; ++i) sS1[w][(i + 8 * hi) * 32 + col] = acc[i] + bb;
        }
    }
    __syncthreads();

    // ---- phase 6: layer-2 vector GEMM + gated atomic scatter (3 WMMAs) ----
    for (int comp = 0; comp < 3; ++comp) {
        if (lane < 16) {
            int j = lane;
            float* rw = &sRaw[w][j * 64];
            float* v1 = &sV1[w][j * 48];
            float avc = rw[38 + comp];
            h2* tr2 = (h2*)&sA[w][j * 64];
#pragma unroll
            for (int k = 0; k < 8; ++k) {
                h2 t = {(half_t)(rw[41 + 2 * k] * avc), (half_t)(rw[42 + 2 * k] * avc)};
                tr2[k] = t;
            }
#pragma unroll
            for (int q = 0; q < 16; q += 2) {
                h2 t = {(half_t)v1[comp * 16 + q], (half_t)v1[comp * 16 + q + 1]};
                tr2[8 + (q >> 1)] = t;
            }
        }
        __syncthreads();
        v16h A  = load_A_frag(&sA[w][0], lane, 0);
        v8f acc = {};
        acc = wmma16(A, load_B_pre(&sBfrag[8 * 512], lane), acc);
#pragma unroll
        for (int i = 0; i < 8; ++i) {
            int row = i + 8 * hi;
            if (tile0 + row < E) {
                int rcv  = sRecv[w][row];
                float gg = sigmoid_f(sS1[w][row * 32 + lo]);  // layer-2 gates (cols 0..15)
                atomicAdd(&agg_mv[(size_t)rcv * 48 + lo * 3 + comp], acc[i] * gg);
            }
        }
        __syncthreads();
    }

    // ---- phase 7: scalar message / a_v / count scatter ----
    if (lane < 16) {
        int j = lane;
        if (tile0 + j < E) {
            int rcv   = sRecv[w][j];
            float* s1 = &sS1[w][j * 32];
            float* rw = &sRaw[w][j * 64];
#pragma unroll
            for (int k = 0; k < 16; ++k)
                atomicAdd(&agg_ms[(size_t)rcv * 16 + k], gelu_f(s1[16 + k]));
#pragma unroll
            for (int c = 0; c < 3; ++c)
                atomicAdd(&agg_av[(size_t)rcv * 3 + c], rw[38 + c]);
            atomicAdd(&agg_cnt[rcv], 1.0f);
        }
    }
}

// ---------------------------------------------------------------------------
// Node kernel: one thread per node, weights staged in LDS.
// ---------------------------------------------------------------------------
__global__ void __launch_bounds__(128)
node_kernel(const float* __restrict__ nWs, const float* __restrict__ nbs,
            const float* __restrict__ nWv,
            const float* __restrict__ fWs, const float* __restrict__ fbs,
            const float* __restrict__ fWv,
            float* __restrict__ s_cur, float* __restrict__ v_cur,
            const float* __restrict__ agg_ms, const float* __restrict__ agg_mv,
            const float* __restrict__ agg_av, const float* __restrict__ agg_cnt,
            int N, float inv) {
    __shared__ float shWs[2 * 289 * 17];
    __shared__ float shWv[2 * 289];
    __shared__ float shbs[2 * 17];
    __shared__ float shF[16 * 16 + 16 + 1];
    for (int i = threadIdx.x; i < 2 * 289 * 17; i += blockDim.x) shWs[i] = nWs[i];
    for (int i = threadIdx.x; i < 2 * 289; i += blockDim.x)      shWv[i] = nWv[i];
    for (int i = threadIdx.x; i < 2 * 17; i += blockDim.x)       shbs[i] = nbs[i];
    for (int i = threadIdx.x; i < 16 * 16; i += blockDim.x)      shF[i]  = fWs[i];
    if (threadIdx.x < 16) shF[256 + threadIdx.x] = fbs[threadIdx.x];
    if (threadIdx.x == 0) shF[272] = fWv[0];
    __syncthreads();

    int n = blockIdx.x * blockDim.x + threadIdx.x;
    if (n >= N) return;

    float cnt = agg_cnt[n];
    float sc  = inv / fmaxf(cnt, 1.0f);  // segment-mean * message normalization
    float ys[17], yv[17][3];
#pragma unroll
    for (int k = 0; k < 16; ++k) ys[k] = agg_ms[n * 16 + k] * sc;
    ys[16] = cnt * sc;
    for (int q = 0; q < 16; ++q)
#pragma unroll
        for (int d = 0; d < 3; ++d) yv[q][d] = agg_mv[n * 48 + q * 3 + d] * sc;
#pragma unroll
    for (int d = 0; d < 3; ++d) yv[16][d] = agg_av[n * 3 + d] * sc;

    float s0[16], ns[16], v0[3], nv[3];
#pragma unroll
    for (int k = 0; k < 16; ++k) { s0[k] = s_cur[n * 16 + k]; ns[k] = s0[k]; }
#pragma unroll
    for (int d = 0; d < 3; ++d) { v0[d] = v_cur[n * 3 + d]; nv[d] = v0[d]; }

    for (int b = 0; b < 2; ++b) {
        const float* W  = &shWs[b * 289 * 17];
        const float* Wv = &shWv[b * 289];
        float acc[17];
#pragma unroll
        for (int q = 0; q < 17; ++q) acc[q] = shbs[b * 17 + q];
        float vx = 0.0f, vy = 0.0f, vz = 0.0f;
        // ts (0e x 0e) part: p = a*17+cc
        for (int a = 0; a < 16; ++a)
            for (int cc = 0; cc < 17; ++cc) {
                float tsv = ns[a] * ys[cc];
                const float* Wr = &W[(a * 17 + cc) * 17];
#pragma unroll
                for (int q = 0; q < 17; ++q) acc[q] += tsv * Wr[q];
            }
        // ts (1o . 1o) part: p = 272+d
        for (int d = 0; d < 17; ++d) {
            float tsv = nv[0] * yv[d][0] + nv[1] * yv[d][1] + nv[2] * yv[d][2];
            const float* Wr = &W[(272 + d) * 17];
#pragma unroll
            for (int q = 0; q < 17; ++q) acc[q] += tsv * Wr[q];
        }
        // tv (0e x 1o) part
        for (int a = 0; a < 16; ++a)
            for (int d = 0; d < 17; ++d) {
                float f = ns[a] * Wv[a * 17 + d];
                vx += f * yv[d][0]; vy += f * yv[d][1]; vz += f * yv[d][2];
            }
        // tv (1o x 0e) part
        for (int cc = 0; cc < 17; ++cc) {
            float f = ys[cc] * Wv[272 + cc];
            vx += f * nv[0]; vy += f * nv[1]; vz += f * nv[2];
        }
        float gate = sigmoid_f(acc[0]);
        float nns[16];
#pragma unroll
        for (int k = 0; k < 16; ++k) nns[k] = gelu_f(acc[1 + k]);
#pragma unroll
        for (int k = 0; k < 16; ++k) ns[k] = nns[k];
        nv[0] = vx * gate; nv[1] = vy * gate; nv[2] = vz * gate;
    }

    // final linear back to irreps_in + residual
#pragma unroll
    for (int q = 0; q < 16; ++q) {
        float so = shF[256 + q];
#pragma unroll
        for (int a = 0; a < 16; ++a) so += ns[a] * shF[a * 16 + q];
        s_cur[n * 16 + q] = s0[q] + so;
    }
    float fw = shF[272];
#pragma unroll
    for (int d = 0; d < 3; ++d) v_cur[n * 3 + d] = v0[d] + nv[d] * fw;
}

// ---------------------------------------------------------------------------
extern "C" void kernel_launch(void* const* d_in, const int* in_sizes, int n_in,
                              void* d_out, int out_size, void* d_ws, size_t ws_size,
                              hipStream_t stream) {
    (void)in_sizes; (void)n_in; (void)out_size; (void)ws_size;
    const float* node_scalars = (const float*)d_in[0];
    const float* node_pos     = (const float*)d_in[1];
    const int*   senders      = (const int*)d_in[2];
    const int*   receivers    = (const int*)d_in[3];
    const float* eWs0 = (const float*)d_in[4];
    const float* ebs0 = (const float*)d_in[5];
    const float* eWv0 = (const float*)d_in[6];
    const float* eWs1 = (const float*)d_in[7];
    const float* ebs1 = (const float*)d_in[8];
    const float* eWv1 = (const float*)d_in[9];
    const float* nWs  = (const float*)d_in[10];
    const float* nbs  = (const float*)d_in[11];
    const float* nWv  = (const float*)d_in[12];
    const float* fWs  = (const float*)d_in[13];
    const float* fbs  = (const float*)d_in[14];
    const float* fWv  = (const float*)d_in[15];

    const int N = N_NODES, E = N_EDGES;
    const float inv = 1.0f / (float)(E - 1);

    float* ws      = (float*)d_ws;
    float* s_cur   = ws;                       // N*16
    float* v_cur   = ws + (size_t)N * 16;      // N*3
    float* agg_ms  = ws + (size_t)N * 19;      // N*16
    float* agg_mv  = agg_ms + (size_t)N * 16;  // N*48
    float* agg_av  = agg_mv + (size_t)N * 48;  // N*3
    float* agg_cnt = agg_av + (size_t)N * 3;   // N

    init_state<<<(N * 16 + 255) / 256, 256, 0, stream>>>(node_scalars, node_pos,
                                                         s_cur, v_cur, N);
    for (int t = 0; t < STEPS; ++t) {
        zero_kernel<<<(N * 68 + 255) / 256, 256, 0, stream>>>(agg_ms, N * 68);
        edge_kernel<<<(E + EPB - 1) / EPB, EWAVES * 32, 0, stream>>>(
            s_cur, v_cur, senders, receivers,
            eWs0 + t * 34 * 32, ebs0 + t * 32, eWv0 + t * 34 * 16,
            eWs1 + t * 32 * 32, ebs1 + t * 32, eWv1 + t * 32 * 16,
            agg_ms, agg_mv, agg_av, agg_cnt, E);
        node_kernel<<<(N + 127) / 128, 128, 0, stream>>>(
            nWs + t * 2 * 289 * 17, nbs + t * 2 * 17, nWv + t * 2 * 289,
            fWs + t * 256, fbs + t * 16, fWv + t,
            s_cur, v_cur, agg_ms, agg_mv, agg_av, agg_cnt, N, inv);
    }
    write_out<<<(N + 255) / 256, 256, 0, stream>>>(s_cur, v_cur, (float*)d_out, N);
}
